// ExplainableMultiheadAttention_7327214206978
// MI455X (gfx1250) — matrix operand order repaired and checked by reading, not compiled, not run on hardware
//
#include <hip/hip_runtime.h>
#include <hip/hip_bf16.h>

typedef __attribute__((ext_vector_type(16))) _Float16 v16h;
typedef __attribute__((ext_vector_type(8)))  float    v8f;

#define B_ 2
#define S_ 2048
#define D_ 1024
#define H_ 16
#define HD_ 64
#define SCALE_ 0.125f   // 1/sqrt(64)

// LDS B-panel tile: 64 N-rows x 32 K halves, padded row pitch (80 B) to spread banks.
#define BT_ROW 40       // halves per LDS row: 32 data + 8 pad; 80 bytes, 16B-aligned

// ---------------------------------------------------------------------------
// WMMA fragment loaders (wave32, v_wmma_f32_16x16x32_f16 layouts per ISA 7.12.2)
// ---------------------------------------------------------------------------

// A: 16x32 f16, row-major source with leading dim lda. Per-lane bytes are two
// contiguous 16-byte runs -> 2x global_load_b128.
__device__ __forceinline__ v16h frag_a_f16(const _Float16* __restrict__ A, int lda) {
  const int lane = threadIdx.x & 31;
  const int row  = lane & 15;
  const int hi   = lane >> 4;
  const _Float16* rp = A + (size_t)row * lda;
  v16h a;
#pragma unroll
  for (int v = 0; v < 8; ++v) {
    const int k0 = ((v & 4) << 2) + hi * 8 + ((v & 3) << 1); // (v<4?0:16)+hi*8+2*(v&3)
    a[2 * v]     = rp[k0];
    a[2 * v + 1] = rp[k0 + 1];
  }
  return a;
}

// Same A layout, f32 source (softmaxed probabilities) -> convert inline.
__device__ __forceinline__ v16h frag_a_f32(const float* __restrict__ A, int lda) {
  const int lane = threadIdx.x & 31;
  const int row  = lane & 15;
  const int hi   = lane >> 4;
  const float* rp = A + (size_t)row * lda;
  v16h a;
#pragma unroll
  for (int v = 0; v < 8; ++v) {
    const int k0 = ((v & 4) << 2) + hi * 8 + ((v & 3) << 1);
    a[2 * v]     = (_Float16)rp[k0];
    a[2 * v + 1] = (_Float16)rp[k0 + 1];
  }
  return a;
}

// B fragment from the LDS-staged N-major panel (64 x BT_ROW halves).
// Element (k, n) lives at tile[(n)*BT_ROW + k]; per-lane read is two
// 16-byte runs (k = 8*hi.. and 16+8*hi..) -> 2x ds_load_b128.
__device__ __forceinline__ v16h frag_bt_lds(const _Float16* tile, int n16) {
  const int lane = threadIdx.x & 31;
  const int col  = lane & 15;
  const int hi   = lane >> 4;
  const _Float16* rp = tile + (n16 + col) * BT_ROW + hi * 8;
  v16h b;
#pragma unroll
  for (int v = 0; v < 8; ++v) {
    b[2 * v]     = rp[2 * v + ((v & 4) << 1)];
    b[2 * v + 1] = rp[2 * v + 1 + ((v & 4) << 1)];
  }
  return b;
}

__device__ __forceinline__ v8f wmma_f16(v16h a, v16h b, v8f c) {
  return __builtin_amdgcn_wmma_f32_16x16x32_f16(false, a, false, b, (short)0, c,
                                                false, false);
}

// ---------------------------------------------------------------------------
// Async staging: copy a 64-row x 32-half panel (row stride ldx halves) from a
// N-major global matrix into an LDS tile via GLOBAL_LOAD_ASYNC_TO_LDS_B128
// (ASYNCcnt path, ISA 10.7/15.18.3). 128 threads x 2 chunks of 16 B = 4 KB.
// ---------------------------------------------------------------------------
__device__ __forceinline__ void stage_bt_panel(uint32_t lds_base,
                                               const _Float16* gbase, int ldx) {
  const int t = threadIdx.x;  // 0..127
#pragma unroll
  for (int i = 0; i < 2; ++i) {
    const int c   = 2 * t + i;   // 0..255 16-byte chunks
    const int row = c >> 2;      // 0..63
    const int off = c & 3;       // chunk within row
    const _Float16* src = gbase + (size_t)row * ldx + off * 8;
    const uint32_t  dst = lds_base + (uint32_t)(row * (BT_ROW * 2) + off * 16);
    asm volatile("global_load_async_to_lds_b128 %0, %1, off"
                 :: "v"(dst), "v"(src)
                 : "memory");
  }
}

// Wait for async-DMA: n==0 drain all, n==2 leave the prestaged tile in flight.
__device__ __forceinline__ void wait_async_drain() {
  asm volatile("s_wait_asynccnt 0x0" ::: "memory");
}
__device__ __forceinline__ void wait_async_keep2() {
  asm volatile("s_wait_asynccnt 0x2" ::: "memory");
}

// ---------------------------------------------------------------------------
// f32 -> f16 straight conversion (activations)
// ---------------------------------------------------------------------------
__global__ void cvt_f32_to_f16(_Float16* __restrict__ dst,
                               const float* __restrict__ src, int n) {
  int i = blockIdx.x * 256 + threadIdx.x;
  if (i < n) dst[i] = (_Float16)src[i];
}

// f32 -> f16 transposing conversion (weights): dst[c*rows + r] = src[r*cols + c]
// grid (cols/32, rows/32, Z), block (32, 8)
__global__ void cvt_transpose_f32_f16(const float* __restrict__ src,
                                      _Float16* __restrict__ dst, int rows,
                                      int cols, size_t zStrideS, size_t zStrideD) {
  __shared__ _Float16 tile[32][33];
  const float* s = src + (size_t)blockIdx.z * zStrideS;
  _Float16*    d = dst + (size_t)blockIdx.z * zStrideD;
  const int r0 = blockIdx.y * 32, c0 = blockIdx.x * 32;
  for (int i = threadIdx.y; i < 32; i += 8)
    tile[i][threadIdx.x] = (_Float16)s[(size_t)(r0 + i) * cols + c0 + threadIdx.x];
  __syncthreads();
  for (int i = threadIdx.y; i < 32; i += 8)
    d[(size_t)(c0 + i) * rows + r0 + threadIdx.x] = tile[threadIdx.x][i];
}

// f16 -> f16 transpose (Vp -> V^T per batch), same mapping as above
__global__ void transpose_f16(const _Float16* __restrict__ src,
                              _Float16* __restrict__ dst, int rows, int cols,
                              size_t zStride) {
  __shared__ _Float16 tile[32][33];
  const _Float16* s = src + (size_t)blockIdx.z * zStride;
  _Float16*       d = dst + (size_t)blockIdx.z * zStride;
  const int r0 = blockIdx.y * 32, c0 = blockIdx.x * 32;
  for (int i = threadIdx.y; i < 32; i += 8)
    tile[i][threadIdx.x] = s[(size_t)(r0 + i) * cols + c0 + threadIdx.x];
  __syncthreads();
  for (int i = threadIdx.y; i < 32; i += 8)
    d[(size_t)(c0 + i) * rows + r0 + threadIdx.x] = tile[threadIdx.x][i];
}

// ---------------------------------------------------------------------------
// Batched GEMM + bias with transposed (N-major) B: C[z] = A @ Bt[z]^T + bias[z]
// A: f16 [M x K] row-major; Bt: f16 [N x K] row-major (leading dim ldbt).
// Double-buffered async LDS staging of the 64x32 B panel; each of the 4 waves
// owns a 32x64 C strip (2 A frags x 4 B frags = 8 WMMAs per k-step, each B
// fragment reused twice). grid = (N/64, M/128, Z), block = 128.
// ---------------------------------------------------------------------------
template <bool STORE_F32>
__global__ void gemm_bias_bt(const _Float16* __restrict__ A,
                             const _Float16* __restrict__ Bt,
                             const float* __restrict__ bias,
                             float* __restrict__ Cf32,
                             _Float16* __restrict__ Cf16, int K, int lda,
                             int ldbt, int ldc, size_t btStrideZ,
                             size_t biasStrideZ, size_t cStrideZ) {
  __shared__ _Float16 bt[2][64 * BT_ROW];
  const int wave = threadIdx.x >> 5;
  const int lane = threadIdx.x & 31;
  const int col  = lane & 15;
  const int hi   = lane >> 4;
  const size_t z = blockIdx.z;
  const _Float16* Bz  = Bt + z * btStrideZ;
  const float*    bsz = bias + z * biasStrideZ;
  const int m0 = blockIdx.y * 128 + wave * 32;
  const int n0 = blockIdx.x * 64;
  const uint32_t base0 = (uint32_t)(uintptr_t)(&bt[0][0]);
  const uint32_t base1 = (uint32_t)(uintptr_t)(&bt[1][0]);

  v8f acc[2][4];
#pragma unroll
  for (int rg = 0; rg < 2; ++rg)
#pragma unroll
    for (int j = 0; j < 4; ++j)
#pragma unroll
      for (int r = 0; r < 8; ++r) acc[rg][j][r] = 0.0f;

  const _Float16* A0 = A + (size_t)m0 * lda;
  const _Float16* A1 = A + (size_t)(m0 + 16) * lda;
  const _Float16* Bp = Bz + (size_t)n0 * ldbt;

  stage_bt_panel(base0, Bp, ldbt);
  int buf = 0;
  for (int k0 = 0; k0 < K; k0 += 32, buf ^= 1) {
    const bool hasNext = (k0 + 32 < K);
    if (hasNext) stage_bt_panel(buf ? base0 : base1, Bp + k0 + 32, ldbt);
    __builtin_prefetch((const void*)(A0 + k0 + 32), 0, 0);
    __builtin_prefetch((const void*)(A1 + k0 + 32), 0, 0);
    v16h af0 = frag_a_f16(A0 + k0, lda);
    v16h af1 = frag_a_f16(A1 + k0, lda);
    if (hasNext) wait_async_keep2(); else wait_async_drain();
    __syncthreads();
    const _Float16* cur = &bt[buf][0];
    v16h bf0 = frag_bt_lds(cur, 0);
    v16h bf1 = frag_bt_lds(cur, 16);
    v16h bf2 = frag_bt_lds(cur, 32);
    v16h bf3 = frag_bt_lds(cur, 48);
    acc[0][0] = wmma_f16(af0, bf0, acc[0][0]);
    acc[1][0] = wmma_f16(af1, bf0, acc[1][0]);
    acc[0][1] = wmma_f16(af0, bf1, acc[0][1]);
    acc[1][1] = wmma_f16(af1, bf1, acc[1][1]);
    acc[0][2] = wmma_f16(af0, bf2, acc[0][2]);
    acc[1][2] = wmma_f16(af1, bf2, acc[1][2]);
    acc[0][3] = wmma_f16(af0, bf3, acc[0][3]);
    acc[1][3] = wmma_f16(af1, bf3, acc[1][3]);
    __syncthreads();
  }

#pragma unroll
  for (int rg = 0; rg < 2; ++rg)
#pragma unroll
    for (int j = 0; j < 4; ++j) {
      const int cc = n0 + 16 * j + col;
      const float bv = bsz[cc];
#pragma unroll
      for (int r = 0; r < 8; ++r) {
        const int rr = m0 + rg * 16 + hi * 8 + r;
        const float val = acc[rg][j][r] + bv;
        if (STORE_F32)
          Cf32[z * cStrideZ + (size_t)rr * ldc + cc] = val;
        else
          Cf16[z * cStrideZ + (size_t)rr * ldc + cc] = (_Float16)val;
      }
    }
}

// ---------------------------------------------------------------------------
// Logit GEMM: out[z] = scale * (Aq[slice] @ Bk[slice]^T), f32 direct to d_out.
// Both operands row-major [S x K-slice]; B panel double-buffer async-staged.
// z = (bz = z/ZH, hz = z%ZH); slice offset = bz*batchStrideA + hz*headStrideA.
// grid = (S/64, S/128, Z), block 128.
// ---------------------------------------------------------------------------
__global__ void gemm_logits(const _Float16* __restrict__ Aq,
                            const _Float16* __restrict__ Bk,
                            float* __restrict__ out, int K, int lda, int ZH,
                            size_t batchStrideA, size_t headStrideA,
                            size_t outStrideZ, int ldc, float scale) {
  __shared__ _Float16 bt[2][64 * BT_ROW];
  const int wave = threadIdx.x >> 5;
  const int lane = threadIdx.x & 31;
  const int col  = lane & 15;
  const int hi   = lane >> 4;
  const int z  = blockIdx.z;
  const int bz = z / ZH;
  const int hz = z % ZH;
  const size_t aoff = (size_t)bz * batchStrideA + (size_t)hz * headStrideA;
  const _Float16* Az = Aq + aoff;
  const _Float16* Bz = Bk + aoff;
  const int m0 = blockIdx.y * 128 + wave * 32;
  const int n0 = blockIdx.x * 64;
  const uint32_t base0 = (uint32_t)(uintptr_t)(&bt[0][0]);
  const uint32_t base1 = (uint32_t)(uintptr_t)(&bt[1][0]);

  v8f acc[2][4];
#pragma unroll
  for (int rg = 0; rg < 2; ++rg)
#pragma unroll
    for (int j = 0; j < 4; ++j)
#pragma unroll
      for (int r = 0; r < 8; ++r) acc[rg][j][r] = 0.0f;

  const _Float16* A0 = Az + (size_t)m0 * lda;
  const _Float16* A1 = Az + (size_t)(m0 + 16) * lda;
  const _Float16* Bp = Bz + (size_t)n0 * lda;

  stage_bt_panel(base0, Bp, lda);
  int buf = 0;
  for (int k0 = 0; k0 < K; k0 += 32, buf ^= 1) {
    const bool hasNext = (k0 + 32 < K);
    if (hasNext) stage_bt_panel(buf ? base0 : base1, Bp + k0 + 32, lda);
    __builtin_prefetch((const void*)(A0 + k0 + 32), 0, 0);
    __builtin_prefetch((const void*)(A1 + k0 + 32), 0, 0);
    v16h af0 = frag_a_f16(A0 + k0, lda);
    v16h af1 = frag_a_f16(A1 + k0, lda);
    if (hasNext) wait_async_keep2(); else wait_async_drain();
    __syncthreads();
    const _Float16* cur = &bt[buf][0];
    v16h bf0 = frag_bt_lds(cur, 0);
    v16h bf1 = frag_bt_lds(cur, 16);
    v16h bf2 = frag_bt_lds(cur, 32);
    v16h bf3 = frag_bt_lds(cur, 48);
    acc[0][0] = wmma_f16(af0, bf0, acc[0][0]);
    acc[1][0] = wmma_f16(af1, bf0, acc[1][0]);
    acc[0][1] = wmma_f16(af0, bf1, acc[0][1]);
    acc[1][1] = wmma_f16(af1, bf1, acc[1][1]);
    acc[0][2] = wmma_f16(af0, bf2, acc[0][2]);
    acc[1][2] = wmma_f16(af1, bf2, acc[1][2]);
    acc[0][3] = wmma_f16(af0, bf3, acc[0][3]);
    acc[1][3] = wmma_f16(af1, bf3, acc[1][3]);
    __syncthreads();
  }

  float* oz = out + (size_t)z * outStrideZ;
#pragma unroll
  for (int rg = 0; rg < 2; ++rg)
#pragma unroll
    for (int j = 0; j < 4; ++j) {
      const int cc = n0 + 16 * j + col;
#pragma unroll
      for (int r = 0; r < 8; ++r) {
        const int rr = m0 + rg * 16 + hi * 8 + r;
        oz[(size_t)rr * ldc + cc] = acc[rg][j][r] * scale;
      }
    }
}

// ---------------------------------------------------------------------------
// In-place softmax over rows of length 2048. grid = numRows, block = 256.
// ---------------------------------------------------------------------------
__global__ void softmax_rows2048(float* __restrict__ buf) {
  float* row = buf + (size_t)blockIdx.x * S_;
  const int t    = threadIdx.x;
  const int wave = t >> 5;
  const int lane = t & 31;
  __shared__ float sred[8];

  float vals[8];
  float m = -3.0e38f;
#pragma unroll
  for (int i = 0; i < 8; ++i) {
    vals[i] = row[t + i * 256];
    m = fmaxf(m, vals[i]);
  }
#pragma unroll
  for (int off = 16; off > 0; off >>= 1) m = fmaxf(m, __shfl_xor(m, off, 32));
  if (lane == 0) sred[wave] = m;
  __syncthreads();
  if (wave == 0) {
    float mm = (lane < 8) ? sred[lane] : -3.0e38f;
#pragma unroll
    for (int off = 4; off > 0; off >>= 1) mm = fmaxf(mm, __shfl_xor(mm, off, 32));
    if (lane == 0) sred[0] = mm;
  }
  __syncthreads();
  m = sred[0];
  __syncthreads();

  float s = 0.0f;
#pragma unroll
  for (int i = 0; i < 8; ++i) {
    vals[i] = __expf(vals[i] - m);
    s += vals[i];
  }
#pragma unroll
  for (int off = 16; off > 0; off >>= 1) s += __shfl_xor(s, off, 32);
  if (lane == 0) sred[wave] = s;
  __syncthreads();
  if (wave == 0) {
    float ss = (lane < 8) ? sred[lane] : 0.0f;
#pragma unroll
    for (int off = 4; off > 0; off >>= 1) ss += __shfl_xor(ss, off, 32);
    if (lane == 0) sred[0] = ss;
  }
  __syncthreads();
  const float inv = 1.0f / sred[0];
#pragma unroll
  for (int i = 0; i < 8; ++i) row[t + i * 256] = vals[i] * inv;
}

// ---------------------------------------------------------------------------
// ctx[b, m, h*64+n] = sum_k P[(b*H+h), m, k] * V[b, k, h*64+n]
// P f32 (softmaxed weights in d_out), converted to f16 fragment-inline.
// V supplied as per-batch transpose Vt[b*D + d][S] so the B panel is N-major
// and async-staged like everywhere else. grid = (1, S/128, B*H), block 128.
// ---------------------------------------------------------------------------
__global__ void gemm_ctx(const float* __restrict__ P,
                         const _Float16* __restrict__ Vt,
                         _Float16* __restrict__ ctx) {
  __shared__ _Float16 bt[2][64 * BT_ROW];
  const int wave = threadIdx.x >> 5;
  const int lane = threadIdx.x & 31;
  const int col  = lane & 15;
  const int hi   = lane >> 4;
  const int z = blockIdx.z;
  const int b = z / H_;
  const int h = z % H_;
  const float*    Pz  = P + (size_t)z * S_ * S_;
  const _Float16* Vtz = Vt + (size_t)(b * D_ + h * HD_) * S_;
  _Float16*       Cz  = ctx + (size_t)b * S_ * D_ + h * HD_;
  const int m0 = blockIdx.y * 128 + wave * 32;
  const uint32_t base0 = (uint32_t)(uintptr_t)(&bt[0][0]);
  const uint32_t base1 = (uint32_t)(uintptr_t)(&bt[1][0]);

  v8f acc[2][4];
#pragma unroll
  for (int rg = 0; rg < 2; ++rg)
#pragma unroll
    for (int j = 0; j < 4; ++j)
#pragma unroll
      for (int r = 0; r < 8; ++r) acc[rg][j][r] = 0.0f;

  const float* P0 = Pz + (size_t)m0 * S_;
  const float* P1 = Pz + (size_t)(m0 + 16) * S_;

  stage_bt_panel(base0, Vtz, S_);
  int buf = 0;
  for (int k0 = 0; k0 < S_; k0 += 32, buf ^= 1) {
    const bool hasNext = (k0 + 32 < S_);
    if (hasNext) stage_bt_panel(buf ? base0 : base1, Vtz + k0 + 32, S_);
    __builtin_prefetch((const void*)(P0 + k0 + 32), 0, 0);
    __builtin_prefetch((const void*)(P1 + k0 + 32), 0, 0);
    v16h af0 = frag_a_f32(P0 + k0, S_);
    v16h af1 = frag_a_f32(P1 + k0, S_);
    if (hasNext) wait_async_keep2(); else wait_async_drain();
    __syncthreads();
    const _Float16* cur = &bt[buf][0];
    v16h bf0 = frag_bt_lds(cur, 0);
    v16h bf1 = frag_bt_lds(cur, 16);
    v16h bf2 = frag_bt_lds(cur, 32);
    v16h bf3 = frag_bt_lds(cur, 48);
    acc[0][0] = wmma_f16(af0, bf0, acc[0][0]);
    acc[1][0] = wmma_f16(af1, bf0, acc[1][0]);
    acc[0][1] = wmma_f16(af0, bf1, acc[0][1]);
    acc[1][1] = wmma_f16(af1, bf1, acc[1][1]);
    acc[0][2] = wmma_f16(af0, bf2, acc[0][2]);
    acc[1][2] = wmma_f16(af1, bf2, acc[1][2]);
    acc[0][3] = wmma_f16(af0, bf3, acc[0][3]);
    acc[1][3] = wmma_f16(af1, bf3, acc[1][3]);
    __syncthreads();
  }

#pragma unroll
  for (int rg = 0; rg < 2; ++rg)
#pragma unroll
    for (int j = 0; j < 4; ++j) {
      const int cc = 16 * j + col;
#pragma unroll
      for (int r = 0; r < 8; ++r) {
        const int rr = m0 + rg * 16 + hi * 8 + r;
        Cz[(size_t)rr * D_ + cc] = (_Float16)acc[rg][j][r];
      }
    }
}

// ---------------------------------------------------------------------------
// Host launch
// ---------------------------------------------------------------------------
extern "C" void kernel_launch(void* const* d_in, const int* in_sizes, int n_in,
                              void* d_out, int out_size, void* d_ws,
                              size_t ws_size, hipStream_t stream) {
  const float* query = (const float*)d_in[0];
  const float* key   = (const float*)d_in[1];
  const float* value = (const float*)d_in[2];
  const float* Wq = (const float*)d_in[3];
  const float* bq = (const float*)d_in[4];
  const float* Wk = (const float*)d_in[5];
  const float* bk = (const float*)d_in[6];
  const float* Wv = (const float*)d_in[7];
  const float* bv = (const float*)d_in[8];
  const float* Wo = (const float*)d_in[9];
  const float* bo = (const float*)d_in[10];
  const float* Wa = (const float*)d_in[11];
  const float* ba = (const float*)d_in[12];

  const size_t BSD = (size_t)B_ * S_ * D_;   // 4,194,304
  const size_t DD  = (size_t)D_ * D_;        // 1,048,576
  const size_t SS  = (size_t)S_ * S_;        // 4,194,304

  // d_out layout: attn_output | attn_weights | patterns
  float* out_attn = (float*)d_out;
  float* out_w    = out_attn + BSD;
  float* out_pat  = out_w + (size_t)B_ * H_ * SS;

  // workspace carve-up (f16), ~132 MB total
  _Float16* ws = (_Float16*)d_ws;
  size_t off = 0;
  auto take = [&](size_t n) { _Float16* p = ws + off; off += n; return p; };
  _Float16* qh   = take(BSD);
  _Float16* kh   = take(BSD);
  _Float16* vh   = take(BSD);
  _Float16* Wqt  = take(DD);      // transposed (N-major) weights
  _Float16* Wkt  = take(DD);
  _Float16* Wvt  = take(DD);
  _Float16* Wot  = take(DD);
  _Float16* Wat  = take(3 * DD);
  _Float16* Qp   = take(BSD);
  _Float16* Kp   = take(BSD);
  _Float16* Vp   = take(BSD);
  _Float16* Vt   = take(BSD);     // per-batch transpose of Vp
  _Float16* ctxh = take(BSD);
  _Float16* qasp = take(3 * BSD);
  _Float16* kasp = take(3 * BSD);

  dim3 blk(128, 1, 1);
  dim3 tblk(32, 8, 1);
  auto cgrid = [](size_t n) { return dim3((unsigned)((n + 255) / 256), 1, 1); };

  // 1) conversions: activations straight, weights transposed to N-major
  cvt_f32_to_f16<<<cgrid(BSD), 256, 0, stream>>>(qh, query, (int)BSD);
  cvt_f32_to_f16<<<cgrid(BSD), 256, 0, stream>>>(kh, key, (int)BSD);
  cvt_f32_to_f16<<<cgrid(BSD), 256, 0, stream>>>(vh, value, (int)BSD);
  {
    dim3 g(D_ / 32, D_ / 32, 1);
    cvt_transpose_f32_f16<<<g, tblk, 0, stream>>>(Wq, Wqt, D_, D_, 0, 0);
    cvt_transpose_f32_f16<<<g, tblk, 0, stream>>>(Wk, Wkt, D_, D_, 0, 0);
    cvt_transpose_f32_f16<<<g, tblk, 0, stream>>>(Wv, Wvt, D_, D_, 0, 0);
    cvt_transpose_f32_f16<<<g, tblk, 0, stream>>>(Wo, Wot, D_, D_, 0, 0);
    dim3 ga(D_ / 32, D_ / 32, 3);
    cvt_transpose_f32_f16<<<ga, tblk, 0, stream>>>(Wa, Wat, D_, D_, DD, DD);
  }

  // 2) Q/K/V projections: [4096 x 1024] @ Wt^T + bias -> f16
  {
    dim3 grid(D_ / 64, (B_ * S_) / 128, 1);
    gemm_bias_bt<false><<<grid, blk, 0, stream>>>(qh, Wqt, bq, nullptr, Qp,
                                                  D_, D_, D_, D_, 0, 0, 0);
    gemm_bias_bt<false><<<grid, blk, 0, stream>>>(kh, Wkt, bk, nullptr, Kp,
                                                  D_, D_, D_, D_, 0, 0, 0);
    gemm_bias_bt<false><<<grid, blk, 0, stream>>>(vh, Wvt, bv, nullptr, Vp,
                                                  D_, D_, D_, D_, 0, 0, 0);
  }
  // V^T per batch (row b*D+d of Vt = column d of batch b)
  {
    dim3 g(D_ / 32, S_ / 32, B_);
    transpose_f16<<<g, tblk, 0, stream>>>(Vp, Vt, S_, D_, (size_t)S_ * D_);
  }

  // 3) aspect projections (z = aspect 0..2): q_asp / k_asp -> f16
  {
    dim3 grid(D_ / 64, (B_ * S_) / 128, 3);
    gemm_bias_bt<false><<<grid, blk, 0, stream>>>(qh, Wat, ba, nullptr, qasp,
                                                  D_, D_, D_, D_, DD, D_, BSD);
    gemm_bias_bt<false><<<grid, blk, 0, stream>>>(kh, Wat, ba, nullptr, kasp,
                                                  D_, D_, D_, D_, DD, D_, BSD);
  }

  // 4) attention logits -> d_out (scaled), z = b*H + h, K = 64
  {
    dim3 grid(S_ / 64, S_ / 128, B_ * H_);
    gemm_logits<<<grid, blk, 0, stream>>>(Qp, Kp, out_w, HD_, D_, H_,
                                          BSD / B_, (size_t)HD_, SS, S_, SCALE_);
  }
  // pattern logits -> d_out (scaled), z = a*B + b, K = 1024
  {
    dim3 grid(S_ / 64, S_ / 128, 3 * B_);
    gemm_logits<<<grid, blk, 0, stream>>>(qasp, kasp, out_pat, D_, D_, 1,
                                          (size_t)S_ * D_, 0, SS, S_, SCALE_);
  }

  // 5) in-place softmax over last dim
  softmax_rows2048<<<dim3(B_ * H_ * S_), 256, 0, stream>>>(out_w);
  softmax_rows2048<<<dim3(3 * B_ * S_), 256, 0, stream>>>(out_pat);

  // 6) ctx = P @ V (per b,h) -> f16 [B,S,D]
  {
    dim3 grid(1, S_ / 128, B_ * H_);
    gemm_ctx<<<grid, blk, 0, stream>>>(out_w, Vt, ctxh);
  }

  // 7) attn_output = ctx @ Wo + bo -> f32 d_out
  {
    dim3 grid(D_ / 64, (B_ * S_) / 128, 1);
    gemm_bias_bt<true><<<grid, blk, 0, stream>>>(ctxh, Wot, bo, out_attn,
                                                 nullptr, D_, D_, D_, D_, 0, 0, 0);
  }
}
